// WaterMixHead_19086834663459
// MI455X (gfx1250) — matrix operand order repaired and checked by reading, not compiled, not run on hardware
//
#include <hip/hip_runtime.h>
#include <hip/hip_bf16.h>
#include <math.h>

typedef __attribute__((ext_vector_type(16))) _Float16 v16h;
typedef __attribute__((ext_vector_type(8)))  float    v8f;

// ---------------------------------------------------------------------------
// WMMA helpers (CDNA5 / gfx1250, wave32): D = A(16x32 f16) * B(32x16 f16) + C(f32)
// ---------------------------------------------------------------------------
__device__ __forceinline__ v8f wmma_f16(v16h a, v16h b, v8f c) {
    // (neg_a, A, neg_b, B, c_mod, C, reuse_a, reuse_b)
    return __builtin_amdgcn_wmma_f32_16x16x32_f16(false, a, false, b, (short)0, c,
                                                  false, false);
}

// A operand: 16x32 f16 tile, row-major in LDS with K contiguous (ld in halves).
// ISA layout: lane L<16 -> row L, K {0..7}U{16..23}; lane L>=16 -> row L-16,
// K {8..15}U{24..31}.  Two 16-byte LDS loads per lane.
__device__ __forceinline__ v16h load_tileA(const _Float16* p, int ld) {
    const int lane = threadIdx.x & 31;
    const int row  = lane & 15;
    const int k0   = (lane >> 4) << 3;            // 0 or 8
    const _Float16* r0 = p + row * ld;
    union { v16h h; float4 f[2]; } u;
    u.f[0] = *(const float4*)(r0 + k0);
    u.f[1] = *(const float4*)(r0 + k0 + 16);
    return u.h;
}

// B operand: 32x16 f16 tile, stored TRANSPOSED in LDS (rows = N, cols = K
// contiguous). ISA layout: lanes 0-15 hold K {0..7}U{16..23} of column N=lane,
// lanes 16-31 hold K {8..15}U{24..31} of column N=lane-16.
__device__ __forceinline__ v16h load_tileB(const _Float16* pT, int ld) {
    const int lane = threadIdx.x & 31;
    const int row  = lane & 15;                   // N index
    const int k0   = (lane >> 4) << 3;            // 0 or 8
    const _Float16* r0 = pT + row * ld + k0;
    union { v16h h; float4 f[2]; } u;
    u.f[0] = *(const float4*)(r0);
    u.f[1] = *(const float4*)(r0 + 16);
    return u.h;
}

// D/C layout: lane<16 -> col=lane, rows 0..7 in d[0..7]; lane>=16 -> col=lane-16,
// rows 8..15.
__device__ __forceinline__ void store_D_rowmajor_h(_Float16* base, int ld,
                                                   int tr, int tc, v8f d) {
    const int lane = threadIdx.x & 31;
    const int col  = tc * 16 + (lane & 15);
    const int r0   = tr * 16 + ((lane >> 4) << 3);
#pragma unroll
    for (int r = 0; r < 8; ++r) base[(r0 + r) * ld + col] = (_Float16)d[r];
}

// ---------------------------------------------------------------------------
// Fused kernel: one block (128 thr = 4 waves) per batch element.
// ---------------------------------------------------------------------------
__global__ __launch_bounds__(128)
void WaterMixHead_19086834663459_kernel(const float* __restrict__ soil,
                                        const float* __restrict__ precip,
                                        const float* __restrict__ ck,
                                        const float* __restrict__ cb,
                                        const float* __restrict__ wq,
                                        const float* __restrict__ wk,
                                        const float* __restrict__ wv,
                                        float* __restrict__ out) {
    __shared__ __align__(16) float    xf[33][64];      // x rows 0..31, precip row 32
    __shared__ __align__(16) _Float16 xth[64][32];     // x^T  (A for q/k/v GEMMs)
    __shared__ __align__(16) _Float16 wT[3][32][32];   // wq^T, wk^T, wv^T (B operands)
    __shared__ __align__(16) _Float16 qh[64][32];      // q    (A for QK^T)
    __shared__ __align__(16) _Float16 kh[64][32];      // k    (== B^T for QK^T)
    __shared__ __align__(16) _Float16 vth[32][64];     // v^T  (B^T for attn@v)
    __shared__ __align__(16) float    sbuf[64][64];    // logits f32
    __shared__ __align__(16) _Float16 ah[64][64];      // attn f16 (A for attn@v)

    const int b    = blockIdx.x;
    const int tid  = threadIdx.x;
    const int lane = tid & 31;
    const int wid  = tid >> 5;
    const float scale = 0.17677669529663687f;          // 1/sqrt(32)

    // hint the tiny shared tables into cache (global_prefetch_b8)
    if (tid == 0) {
#pragma unroll
        for (int s = 0; s < 3; ++s) {
            __builtin_prefetch(wq + s * 1024, 0, 0);
            __builtin_prefetch(wk + s * 1024, 0, 0);
            __builtin_prefetch(wv + s * 1024, 0, 0);
            __builtin_prefetch(ck + s * 128, 0, 0);
            __builtin_prefetch(cb + s * 64, 0, 0);
        }
    }

    // ---- load x [32][64] and precip row (vectorized) ----
    const float4* xg4 = (const float4*)(soil + (size_t)b * 2048);
#pragma unroll
    for (int t = 0; t < 4; ++t) {
        int q = tid + t * 128;                       // float4 index, 512 total
        int m = q >> 4, n = (q & 15) * 4;
        *(float4*)&xf[m][n] = xg4[q];
    }
    if (tid < 16)
        *(float4*)&xf[32][tid * 4] =
            ((const float4*)(precip + (size_t)b * 64))[tid];
    __syncthreads();

    for (int step = 0; step < 3; ++step) {
        // ---- vertical mix (depthwise conv, kernel=2, valid), float4 ----
        const float* k0g = ck + (step * 2 + 0) * 64;
        const float* k1g = ck + (step * 2 + 1) * 64;
        const float* bbg = cb + step * 64;
        float4 cv4[4];
#pragma unroll
        for (int t = 0; t < 4; ++t) {
            int q = tid + t * 128;
            int m = q >> 4, n = (q & 15) * 4;
            float4 a  = *(const float4*)&xf[m][n];
            float4 bN = *(const float4*)&xf[m + 1][n];
            float4 K0 = *(const float4*)&k0g[n];
            float4 K1 = *(const float4*)&k1g[n];
            float4 Bb = *(const float4*)&bbg[n];
            cv4[t] = make_float4(a.x * K0.x + bN.x * K1.x + Bb.x,
                                 a.y * K0.y + bN.y * K1.y + Bb.y,
                                 a.z * K0.z + bN.z * K1.z + Bb.z,
                                 a.w * K0.w + bN.w * K1.w + Bb.w);
        }
        __syncthreads();
#pragma unroll
        for (int t = 0; t < 4; ++t) {
            int q = tid + t * 128;
            int m = q >> 4, n = (q & 15) * 4;
            *(float4*)&xf[m][n] = cv4[t];
        }
        // load this step's weights, transposed + f16 (L2-resident after block 0)
        const float* wqg = wq + step * 1024;
        const float* wkg = wk + step * 1024;
        const float* wvg = wv + step * 1024;
#pragma unroll
        for (int t = 0; t < 2; ++t) {
            int q = tid + t * 128;                   // float4 index, 256 total
            int r = q >> 3, c = (q & 7) * 4;         // global w[r][c..c+3]
            float4 aq = ((const float4*)wqg)[q];
            float4 ak = ((const float4*)wkg)[q];
            float4 av = ((const float4*)wvg)[q];
            wT[0][c + 0][r] = (_Float16)aq.x;  wT[0][c + 1][r] = (_Float16)aq.y;
            wT[0][c + 2][r] = (_Float16)aq.z;  wT[0][c + 3][r] = (_Float16)aq.w;
            wT[1][c + 0][r] = (_Float16)ak.x;  wT[1][c + 1][r] = (_Float16)ak.y;
            wT[1][c + 2][r] = (_Float16)ak.z;  wT[1][c + 3][r] = (_Float16)ak.w;
            wT[2][c + 0][r] = (_Float16)av.x;  wT[2][c + 1][r] = (_Float16)av.y;
            wT[2][c + 2][r] = (_Float16)av.z;  wT[2][c + 3][r] = (_Float16)av.w;
        }
        __syncthreads();

        // ---- build x^T in f16 (vector loads, scalar transposed stores) ----
#pragma unroll
        for (int t = 0; t < 4; ++t) {
            int q = tid + t * 128;
            int m = q >> 4, n = (q & 15) * 4;
            float4 v = *(const float4*)&xf[m][n];
            xth[n + 0][m] = (_Float16)v.x;
            xth[n + 1][m] = (_Float16)v.y;
            xth[n + 2][m] = (_Float16)v.z;
            xth[n + 3][m] = (_Float16)v.w;
        }
        __syncthreads();

        // ---- q/k/v = x^T @ w  (wave wid owns row-tile wid; 2 col-tiles each) ----
        {
            v16h a = load_tileA(&xth[wid * 16][0], 32);
#pragma unroll
            for (int c = 0; c < 2; ++c) {
                v8f dq = {};
                dq = wmma_f16(a, load_tileB(&wT[0][c * 16][0], 32), dq);
                store_D_rowmajor_h(&qh[0][0], 32, wid, c, dq);

                v8f dk = {};
                dk = wmma_f16(a, load_tileB(&wT[1][c * 16][0], 32), dk);
                store_D_rowmajor_h(&kh[0][0], 32, wid, c, dk);

                v8f dv = {};
                dv = wmma_f16(a, load_tileB(&wT[2][c * 16][0], 32), dv);
                // store v transposed: vth[mfeat][ndiv]  (contiguous in r)
                int vr  = c * 16 + (lane & 15);
                int nc0 = wid * 16 + ((lane >> 4) << 3);
#pragma unroll
                for (int r = 0; r < 8; ++r) vth[vr][nc0 + r] = (_Float16)dv[r];
            }
        }
        __syncthreads();

        // ---- logits = q @ k^T * scale  (kh rows are already B^T rows) ----
        {
            v16h a = load_tileA(&qh[wid * 16][0], 32);
#pragma unroll
            for (int c = 0; c < 4; ++c) {
                v8f d = {};
                d = wmma_f16(a, load_tileB(&kh[c * 16][0], 32), d);
                int col = c * 16 + (lane & 15);
                int r0  = wid * 16 + ((lane >> 4) << 3);
#pragma unroll
                for (int r = 0; r < 8; ++r) sbuf[r0 + r][col] = d[r] * scale;
            }
        }
        __syncthreads();

        // ---- softmax: 128 threads, half-row (32 elems) each; the two halves
        //      of a row live in adjacent lanes -> combine with shfl_xor(1) ----
        {
            const int row = tid >> 1;
            const int h32 = (tid & 1) * 32;
            const float4* rp = (const float4*)&sbuf[row][h32];
            float4 vb[8];
            float mx = -1e30f;
#pragma unroll
            for (int j = 0; j < 8; ++j) {
                vb[j] = rp[j];
                mx = fmaxf(mx, fmaxf(fmaxf(vb[j].x, vb[j].y),
                                     fmaxf(vb[j].z, vb[j].w)));
            }
            mx = fmaxf(mx, __shfl_xor(mx, 1));
            float sum = 0.f;
#pragma unroll
            for (int j = 0; j < 8; ++j) {
                vb[j].x = __expf(vb[j].x - mx);
                vb[j].y = __expf(vb[j].y - mx);
                vb[j].z = __expf(vb[j].z - mx);
                vb[j].w = __expf(vb[j].w - mx);
                sum += (vb[j].x + vb[j].y) + (vb[j].z + vb[j].w);
            }
            sum += __shfl_xor(sum, 1);
            const float inv = 1.f / sum;
            _Float16* dst = &ah[row][h32];
#pragma unroll
            for (int j = 0; j < 4; ++j) {
                union { _Float16 h[8]; float4 f; } pk;
                float4 lo = vb[2 * j], hi = vb[2 * j + 1];
                pk.h[0] = (_Float16)(lo.x * inv); pk.h[1] = (_Float16)(lo.y * inv);
                pk.h[2] = (_Float16)(lo.z * inv); pk.h[3] = (_Float16)(lo.w * inv);
                pk.h[4] = (_Float16)(hi.x * inv); pk.h[5] = (_Float16)(hi.y * inv);
                pk.h[6] = (_Float16)(hi.z * inv); pk.h[7] = (_Float16)(hi.w * inv);
                *(float4*)(dst + 8 * j) = pk.f;
            }
        }
        __syncthreads();

        // ---- x^T = attn @ v   (K=64 -> two WMMAs), write back as x (+ReLU) ----
        {
            v16h a0 = load_tileA(&ah[wid * 16][0], 64);
            v16h a1 = load_tileA(&ah[wid * 16][32], 64);
            const bool dorelu = (step < 2);
#pragma unroll
            for (int c = 0; c < 2; ++c) {
                v8f d = {};
                d = wmma_f16(a0, load_tileB(&vth[c * 16][0], 64), d);
                d = wmma_f16(a1, load_tileB(&vth[c * 16][32], 64), d);
                int mrow = c * 16 + (lane & 15);       // feature (= x row)
                int nc0  = wid * 16 + ((lane >> 4) << 3);
#pragma unroll
                for (int r = 0; r < 8; ++r) {
                    float v = d[r];
                    if (dorelu) v = fmaxf(v, 0.f);
                    xf[mrow][nc0 + r] = v;
                }
            }
        }
        __syncthreads();
    }

    // ---- tanh + store (vectorized) ----
    float4* og4 = (float4*)(out + (size_t)b * 2048);
#pragma unroll
    for (int t = 0; t < 4; ++t) {
        int q = tid + t * 128;
        int m = q >> 4, n = (q & 15) * 4;
        float4 v = *(const float4*)&xf[m][n];
        og4[q] = make_float4(tanhf(v.x), tanhf(v.y), tanhf(v.z), tanhf(v.w));
    }
}

extern "C" void kernel_launch(void* const* d_in, const int* in_sizes, int n_in,
                              void* d_out, int out_size, void* d_ws, size_t ws_size,
                              hipStream_t stream) {
    (void)n_in; (void)d_ws; (void)ws_size; (void)out_size;
    const float* soil   = (const float*)d_in[0];
    const float* precip = (const float*)d_in[1];
    const float* ck     = (const float*)d_in[2];
    const float* cb     = (const float*)d_in[3];
    const float* wq     = (const float*)d_in[4];
    const float* wk     = (const float*)d_in[5];
    const float* wv     = (const float*)d_in[6];
    float* out = (float*)d_out;

    const int B = in_sizes[0] / (32 * 64);   // 16384
    hipLaunchKernelGGL(WaterMixHead_19086834663459_kernel, dim3(B), dim3(128), 0,
                       stream, soil, precip, ck, cb, wq, wk, wv, out);
}
